// MultiHeadAttention_38817914421639
// MI455X (gfx1250) — compile-verified, matrix operand order checked by reading, not run on hardware
//
#include <hip/hip_runtime.h>
#include <hip/hip_bf16.h>

// ---------------- constants ----------------
#define BB 2
#define SS 2048
#define DD 1024
#define HH 16
#define DEPTH 64
#define MM (BB * SS)      // 4096 rows for all GEMMs

typedef __attribute__((ext_vector_type(16))) __bf16 v16bf;
typedef __attribute__((ext_vector_type(8)))  float  v8f;
typedef __attribute__((ext_vector_type(4)))  unsigned int u32x4;

union FragBF { u32x4 q[2]; v16bf v; };

static __device__ __forceinline__ v8f wmma_bf16(const FragBF& a, const FragBF& b, v8f c) {
  return __builtin_amdgcn_wmma_f32_16x16x32_bf16(false, a.v, false, b.v, (short)0, c, false, false);
}

static __device__ __forceinline__ float bf2f(unsigned short h) {
  unsigned int u = ((unsigned int)h) << 16;
  float f; __builtin_memcpy(&f, &u, 4); return f;
}
static __device__ __forceinline__ unsigned short f2bf(float f) {
  unsigned int u; __builtin_memcpy(&u, &f, 4);
  unsigned int r = u + 0x7FFFu + ((u >> 16) & 1u);   // round-to-nearest-even
  return (unsigned short)(r >> 16);
}
static __device__ __forceinline__ float silu(float x) {
  return x / (1.0f + __expf(-x));
}

// ---------------- f32 -> bf16 convert ----------------
__global__ __launch_bounds__(256) void cvt_bf16_kernel(const float* __restrict__ in,
                                                       unsigned short* __restrict__ out, int n) {
  int i = blockIdx.x * 256 + threadIdx.x;
  if (i < n) out[i] = f2bf(in[i]);
}

// ---------------- weight transpose+convert: W[K][N] f32 -> Wt[N][K] bf16 ----------------
__global__ __launch_bounds__(256) void tcvt_kernel(const float* __restrict__ W,
                                                   unsigned short* __restrict__ Wt) {
  int i = blockIdx.x * 256 + threadIdx.x;     // DD*DD elements
  int k = i & (DD - 1);
  int n = i >> 10;
  Wt[n * DD + k] = f2bf(W[k * DD + n]);
}

// ---------------- depthwise conv1d, K=3, SAME, bf16 in/out ----------------
__global__ __launch_bounds__(256) void dwconv_kernel(const unsigned short* __restrict__ x,
                                                     const float* __restrict__ dw,
                                                     unsigned short* __restrict__ y) {
  int idx = blockIdx.x * 256 + threadIdx.x;   // B*S*D
  int d = idx & (DD - 1);
  int s = (idx >> 10) & (SS - 1);
  int b = idx >> 21;
  float acc = 0.0f;
#pragma unroll
  for (int t = 0; t < 3; ++t) {
    int ss = s + t - 1;
    if (ss >= 0 && ss < SS)
      acc += bf2f(x[((size_t)(b * SS + ss) << 10) + d]) * dw[t * DD + d];
  }
  y[idx] = f2bf(acc);
}

// ---------------- WMMA GEMM: out = silu(A[M,K] @ Wt[N,K]^T + bias) ----------------
// M=4096, N=1024, K=1024. Block 256 thr = 8 waves, tile 128x128, waves 4(M)x2(N),
// each wave computes a 32x64 strip = 8 accumulators. LDS ping-pong double buffer,
// one barrier per k-step, global loads for k+1 overlapped with WMMA chain on k.
__global__ __launch_bounds__(256) void gemm_bias_silu_kernel(
    const unsigned short* __restrict__ A, const unsigned short* __restrict__ Wt,
    const float* __restrict__ bias,
    unsigned short* __restrict__ outb, float* __restrict__ outf) {
  __shared__ __align__(16) unsigned short At[2][128 * 32];
  __shared__ __align__(16) unsigned short Bt[2][128 * 32];

  const int tid  = threadIdx.x;
  const int lane = tid & 31;
  const int wid  = tid >> 5;
  const int wm   = wid & 3;          // 0..3 -> 32-row strip
  const int wn   = wid >> 2;         // 0..1 -> 64-col strip
  const int lh   = lane & 15;
  const int ksel = (lane >> 4) << 3; // ISA A/B lane K pattern: 0 or 8
  const int m0 = blockIdx.x * 128;
  const int n0 = blockIdx.y * 128;

  // loader: each tile is 128x32 ushort = 512 16B chunks; 2 chunks/thread
  const int r0 = tid >> 2;
  const int g0 = (tid & 3) << 3;
  const int r1 = r0 + 64;

  v8f acc[2][4];
#pragma unroll
  for (int p = 0; p < 2; ++p)
#pragma unroll
    for (int f = 0; f < 4; ++f)
#pragma unroll
      for (int j = 0; j < 8; ++j) acc[p][f][j] = 0.0f;

  // prologue: tile 0 -> buffer 0
  *(u32x4*)&At[0][r0 * 32 + g0] = *(const u32x4*)&A [(size_t)(m0 + r0) * DD + g0];
  *(u32x4*)&At[0][r1 * 32 + g0] = *(const u32x4*)&A [(size_t)(m0 + r1) * DD + g0];
  *(u32x4*)&Bt[0][r0 * 32 + g0] = *(const u32x4*)&Wt[(size_t)(n0 + r0) * DD + g0];
  *(u32x4*)&Bt[0][r1 * 32 + g0] = *(const u32x4*)&Wt[(size_t)(n0 + r1) * DD + g0];
  __syncthreads();

  const int NK = DD / 32;            // 32 k-steps
  for (int i = 0; i < NK; ++i) {
    const int cur = i & 1, nxt = cur ^ 1;

    // issue global loads for next tile (latency hidden behind WMMA chain)
    u32x4 ta0, ta1, tb0, tb1;
    const int kb = (i + 1) * 32;
    if (i + 1 < NK) {
      ta0 = *(const u32x4*)&A [(size_t)(m0 + r0) * DD + kb + g0];
      ta1 = *(const u32x4*)&A [(size_t)(m0 + r1) * DD + kb + g0];
      tb0 = *(const u32x4*)&Wt[(size_t)(n0 + r0) * DD + kb + g0];
      tb1 = *(const u32x4*)&Wt[(size_t)(n0 + r1) * DD + kb + g0];
    }
    if (i + 2 < NK) {                // gfx1250 global_prefetch_b8 for k+2 tile
      __builtin_prefetch(&A [(size_t)(m0 + r0) * DD + (i + 2) * 32 + g0], 0, 1);
      __builtin_prefetch(&Wt[(size_t)(n0 + r0) * DD + (i + 2) * 32 + g0], 0, 1);
    }

    // pre-load ALL fragments of this k-step, then run the 8-WMMA chain
    FragBF a[2], b[4];
#pragma unroll
    for (int p = 0; p < 2; ++p) {
      const int ar = wm * 32 + p * 16 + lh;
      a[p].q[0] = *(const u32x4*)&At[cur][ar * 32 + ksel];
      a[p].q[1] = *(const u32x4*)&At[cur][ar * 32 + ksel + 16];
    }
#pragma unroll
    for (int f = 0; f < 4; ++f) {
      const int br = wn * 64 + f * 16 + lh;
      b[f].q[0] = *(const u32x4*)&Bt[cur][br * 32 + ksel];
      b[f].q[1] = *(const u32x4*)&Bt[cur][br * 32 + ksel + 16];
    }
#pragma unroll
    for (int p = 0; p < 2; ++p)
#pragma unroll
      for (int f = 0; f < 4; ++f)
        acc[p][f] = wmma_bf16(a[p], b[f], acc[p][f]);

    // commit next tile to the other buffer
    if (i + 1 < NK) {
      *(u32x4*)&At[nxt][r0 * 32 + g0] = ta0;
      *(u32x4*)&At[nxt][r1 * 32 + g0] = ta1;
      *(u32x4*)&Bt[nxt][r0 * 32 + g0] = tb0;
      *(u32x4*)&Bt[nxt][r1 * 32 + g0] = tb1;
    }
    __syncthreads();
  }

#pragma unroll
  for (int p = 0; p < 2; ++p)
#pragma unroll
    for (int f = 0; f < 4; ++f) {
      const int gc = n0 + wn * 64 + f * 16 + lh;
      const float bv = bias[gc];
#pragma unroll
      for (int j = 0; j < 8; ++j) {
        const int gr = m0 + wm * 32 + p * 16 + ((lane < 16) ? j : j + 8);
        float y = silu(acc[p][f][j] + bv);
        if (outf) outf[(size_t)gr * DD + gc] = y;
        else      outb[(size_t)gr * DD + gc] = f2bf(y);
      }
    }
}

// ---------------- flash attention: per (b,h), 64 query rows/block, 32-key chunks --------
// block = 128 threads = 4 waves, each wave owns 16 query rows.
__global__ __launch_bounds__(128) void attention_kernel(
    const unsigned short* __restrict__ Q, const unsigned short* __restrict__ Kx,
    const unsigned short* __restrict__ V, const int* __restrict__ mask,
    unsigned short* __restrict__ O) {
  __shared__ __align__(16) unsigned short Kc[32 * 64];   // [key][depth]
  __shared__ __align__(16) unsigned short Vt[64 * 32];   // [depth][key]  (V^T)
  __shared__ __align__(16) unsigned short Pt[4 * 16 * 32];

  const int tid  = threadIdx.x;
  const int lane = tid & 31;
  const int wid  = tid >> 5;
  const int lh   = lane & 15;
  const int ksel = (lane >> 4) << 3;
  const int bh = blockIdx.y;
  const int b  = bh >> 4;
  const int h  = bh & 15;
  const int q0 = blockIdx.x * 64 + wid * 16;
  const size_t hoff = (size_t)h * DEPTH;

  // Q fragments: 16 rows x 64 depth -> two A frags (depth split 2x32)
  FragBF aq[2];
  {
    const unsigned short* qp = Q + ((size_t)(b * SS + q0 + lh)) * DD + hoff;
    aq[0].q[0] = *(const u32x4*)&qp[ksel];
    aq[0].q[1] = *(const u32x4*)&qp[ksel + 16];
    aq[1].q[0] = *(const u32x4*)&qp[32 + ksel];
    aq[1].q[1] = *(const u32x4*)&qp[32 + ksel + 16];
  }

  float m[8], l[8];
  v8f o[4];
#pragma unroll
  for (int j = 0; j < 8; ++j) { m[j] = -3.0e38f; l[j] = 0.0f; }
#pragma unroll
  for (int f = 0; f < 4; ++f)
#pragma unroll
    for (int j = 0; j < 8; ++j) o[f][j] = 0.0f;

  const int* mp = mask + b * SS;
  unsigned short* Pw = &Pt[wid * 512];

  for (int c = 0; c < SS / 32; ++c) {
    __syncthreads();
    // K chunk: 32x64 ushort = 256 16B chunks / 128 threads
#pragma unroll
    for (int i = 0; i < 2; ++i) {
      int ch = tid + (i << 7);
      int kr = ch >> 3, cg = (ch & 7) << 3;
      *(u32x4*)&Kc[kr * 64 + cg] =
          *(const u32x4*)&Kx[((size_t)(b * SS + c * 32 + kr)) * DD + hoff + cg];
    }
    // V chunk transposed (scalar stores): Vt[depth][key]
#pragma unroll
    for (int i = 0; i < 16; ++i) {
      int e = tid + (i << 7);                 // 0..2047
      int vk = e >> 6, vd = e & 63;
      Vt[vd * 32 + vk] = V[((size_t)(b * SS + c * 32 + vk)) * DD + hoff + vd];
    }
    __syncthreads();

    // logits: two 16x16 tiles (keys n=0,1), depth 64 = 2 WMMAs each.
    // pre-load all 4 B fragments, then issue the 4-WMMA chain.
    FragBF bk[4];
#pragma unroll
    for (int n = 0; n < 2; ++n) {
      const int br = n * 16 + lh;
#pragma unroll
      for (int kk = 0; kk < 2; ++kk) {
        bk[n * 2 + kk].q[0] = *(const u32x4*)&Kc[br * 64 + kk * 32 + ksel];
        bk[n * 2 + kk].q[1] = *(const u32x4*)&Kc[br * 64 + kk * 32 + ksel + 16];
      }
    }
    v8f s0, s1;
#pragma unroll
    for (int j = 0; j < 8; ++j) { s0[j] = 0.0f; s1[j] = 0.0f; }
    s0 = wmma_bf16(aq[0], bk[0], s0);
    s0 = wmma_bf16(aq[1], bk[1], s0);
    s1 = wmma_bf16(aq[0], bk[2], s1);
    s1 = wmma_bf16(aq[1], bk[3], s1);

    const float pm0 = -1e9f * (float)mp[c * 32 + lh];
    const float pm1 = -1e9f * (float)mp[c * 32 + 16 + lh];
    float corr[8];
#pragma unroll
    for (int j = 0; j < 8; ++j) {
      float a0 = s0[j] * 0.125f + pm0;        // 1/sqrt(64)
      float a1 = s1[j] * 0.125f + pm1;
      float v = fmaxf(a0, a1);
#pragma unroll
      for (int off = 1; off < 16; off <<= 1) v = fmaxf(v, __shfl_xor(v, off, 32));
      float nm = fmaxf(m[j], v);
      corr[j] = __expf(m[j] - nm);
      m[j] = nm;
      float p0 = __expf(a0 - nm);
      float p1 = __expf(a1 - nm);
      float rs = p0 + p1;
#pragma unroll
      for (int off = 1; off < 16; off <<= 1) rs += __shfl_xor(rs, off, 32);
      l[j] = l[j] * corr[j] + rs;
      // stash P (bf16) in per-wave LDS to re-layout as A fragment
      int prow = (lane < 16) ? j : j + 8;
      Pw[prow * 32 + lh]      = f2bf(p0);
      Pw[prow * 32 + 16 + lh] = f2bf(p1);
    }
    __asm__ volatile("" ::: "memory");        // keep DS store->load order (HW in-order per wave)

#pragma unroll
    for (int f = 0; f < 4; ++f)
#pragma unroll
      for (int j = 0; j < 8; ++j) o[f][j] *= corr[j];

    // PV: pre-load P fragment + 4 V fragments, then 4-WMMA chain
    FragBF pa, bv[4];
    pa.q[0] = *(const u32x4*)&Pw[lh * 32 + ksel];
    pa.q[1] = *(const u32x4*)&Pw[lh * 32 + ksel + 16];
#pragma unroll
    for (int f = 0; f < 4; ++f) {
      const int dc = f * 16 + lh;
      bv[f].q[0] = *(const u32x4*)&Vt[dc * 32 + ksel];
      bv[f].q[1] = *(const u32x4*)&Vt[dc * 32 + ksel + 16];
    }
#pragma unroll
    for (int f = 0; f < 4; ++f) o[f] = wmma_bf16(pa, bv[f], o[f]);
  }

#pragma unroll
  for (int f = 0; f < 4; ++f) {
    const int dc = f * 16 + lh;
#pragma unroll
    for (int j = 0; j < 8; ++j) {
      const int gr = q0 + ((lane < 16) ? j : j + 8);
      O[((size_t)(b * SS + gr)) * DD + hoff + dc] = f2bf(o[f][j] / l[j]);
    }
  }
}

// ---------------- host-side orchestration ----------------
extern "C" void kernel_launch(void* const* d_in, const int* in_sizes, int n_in,
                              void* d_out, int out_size, void* d_ws, size_t ws_size,
                              hipStream_t stream) {
  (void)in_sizes; (void)n_in; (void)out_size; (void)ws_size;

  const float* v_in  = (const float*)d_in[0];
  const float* k_in  = (const float*)d_in[1];
  const float* q_in  = (const float*)d_in[2];
  const int*   mask  = (const int*)d_in[3];
  const float* wq_w  = (const float*)d_in[4];
  const float* wq_b  = (const float*)d_in[5];
  const float* wk_w  = (const float*)d_in[6];
  const float* wk_b  = (const float*)d_in[7];
  const float* wv_w  = (const float*)d_in[8];
  const float* wv_b  = (const float*)d_in[9];
  const float* dwq   = (const float*)d_in[10];
  const float* pwq   = (const float*)d_in[11];
  const float* bq    = (const float*)d_in[12];
  const float* dwk   = (const float*)d_in[13];
  const float* pwk   = (const float*)d_in[14];
  const float* bk    = (const float*)d_in[15];
  const float* dwv   = (const float*)d_in[16];
  const float* pwv   = (const float*)d_in[17];
  const float* bv    = (const float*)d_in[18];
  const float* wo_w  = (const float*)d_in[19];
  const float* wo_b  = (const float*)d_in[20];
  float* out = (float*)d_out;

  const size_t ACT = (size_t)MM * DD * sizeof(unsigned short); // 8 MB
  const size_t WMT = (size_t)DD * DD * sizeof(unsigned short); // 2 MB
  char* ws = (char*)d_ws;
  unsigned short* qb  = (unsigned short*)(ws + 0 * ACT);
  unsigned short* kb  = (unsigned short*)(ws + 1 * ACT);
  unsigned short* vb  = (unsigned short*)(ws + 2 * ACT);
  unsigned short* qp  = (unsigned short*)(ws + 3 * ACT);
  unsigned short* kp  = (unsigned short*)(ws + 4 * ACT);
  unsigned short* vp  = (unsigned short*)(ws + 5 * ACT);
  unsigned short* ao  = (unsigned short*)(ws + 6 * ACT);
  char* wbase = ws + 7 * ACT;
  unsigned short* wqT  = (unsigned short*)(wbase + 0 * WMT);
  unsigned short* wkT  = (unsigned short*)(wbase + 1 * WMT);
  unsigned short* wvT  = (unsigned short*)(wbase + 2 * WMT);
  unsigned short* pwqT = (unsigned short*)(wbase + 3 * WMT);
  unsigned short* pwkT = (unsigned short*)(wbase + 4 * WMT);
  unsigned short* pwvT = (unsigned short*)(wbase + 5 * WMT);
  unsigned short* woT  = (unsigned short*)(wbase + 6 * WMT);

  const int NELT = MM * DD;                 // 4,194,304
  const dim3 gCvt((NELT + 255) / 256);
  const dim3 gT((DD * DD) / 256);
  const dim3 gGemm(MM / 128, DD / 128);     // (32, 8)
  const dim3 gAttn(SS / 64, BB * HH);       // (32, 32)

  // 1) inputs -> bf16
  cvt_bf16_kernel<<<gCvt, 256, 0, stream>>>(q_in, qb, NELT);
  cvt_bf16_kernel<<<gCvt, 256, 0, stream>>>(k_in, kb, NELT);
  cvt_bf16_kernel<<<gCvt, 256, 0, stream>>>(v_in, vb, NELT);

  // 2) weights -> [N][K] bf16
  tcvt_kernel<<<gT, 256, 0, stream>>>(wq_w, wqT);
  tcvt_kernel<<<gT, 256, 0, stream>>>(wk_w, wkT);
  tcvt_kernel<<<gT, 256, 0, stream>>>(wv_w, wvT);
  tcvt_kernel<<<gT, 256, 0, stream>>>(pwq, pwqT);
  tcvt_kernel<<<gT, 256, 0, stream>>>(pwk, pwkT);
  tcvt_kernel<<<gT, 256, 0, stream>>>(pwv, pwvT);
  tcvt_kernel<<<gT, 256, 0, stream>>>(wo_w, woT);

  // 3) dense projections + SiLU
  gemm_bias_silu_kernel<<<gGemm, 256, 0, stream>>>(qb, wqT, wq_b, qp, nullptr);
  gemm_bias_silu_kernel<<<gGemm, 256, 0, stream>>>(kb, wkT, wk_b, kp, nullptr);
  gemm_bias_silu_kernel<<<gGemm, 256, 0, stream>>>(vb, wvT, wv_b, vp, nullptr);

  // 4) depthwise conv (reuse qb/kb/vb as outputs)
  dwconv_kernel<<<gCvt, 256, 0, stream>>>(qp, dwq, qb);
  dwconv_kernel<<<gCvt, 256, 0, stream>>>(kp, dwk, kb);
  dwconv_kernel<<<gCvt, 256, 0, stream>>>(vp, dwv, vb);

  // 5) pointwise conv GEMM + SiLU (reuse qp/kp/vp as outputs)
  gemm_bias_silu_kernel<<<gGemm, 256, 0, stream>>>(qb, pwqT, bq, qp, nullptr);
  gemm_bias_silu_kernel<<<gGemm, 256, 0, stream>>>(kb, pwkT, bk, kp, nullptr);
  gemm_bias_silu_kernel<<<gGemm, 256, 0, stream>>>(vb, pwvT, bv, vp, nullptr);

  // 6) flash attention (bf16 in, bf16 out)
  attention_kernel<<<gAttn, 128, 0, stream>>>(qp, kp, vp, mask, ao);

  // 7) output dense + SiLU -> f32 d_out
  gemm_bias_silu_kernel<<<gGemm, 256, 0, stream>>>(ao, woT, wo_b, nullptr, out);
}
